// NodeModel_10892037063287
// MI455X (gfx1250) — matrix operand order repaired and checked by reading, not compiled, not run on hardware
//
#include <hip/hip_runtime.h>

#define NN 50000
#define EE 800000

typedef __attribute__((ext_vector_type(8)))  float  v8f;
typedef __attribute__((ext_vector_type(8)))  __bf16 b8;
typedef __attribute__((ext_vector_type(16))) __bf16 b16v;

// ---------------- helpers ----------------

__device__ __forceinline__ void wave_sync() {
    __builtin_amdgcn_wave_barrier();
    asm volatile("" ::: "memory");
}

// split fp32 into bf16 hi (truncated, exact) + bf16 lo (RNE of residual)
__device__ __forceinline__ void split2(float a, unsigned short& h, unsigned short& l) {
    unsigned u = __float_as_uint(a);
    h = (unsigned short)(u >> 16);
    float hf = __uint_as_float(u & 0xFFFF0000u);
    unsigned v = __float_as_uint(a - hf);
    unsigned r = 0x7FFFu + ((v >> 16) & 1u);
    l = (unsigned short)((v + r) >> 16);
}

__device__ __forceinline__ __bf16 us2bf(unsigned short s) {
    return __builtin_bit_cast(__bf16, s);
}

// build a 16-element bf16 fragment from two contiguous 8-element (16B) loads
__device__ __forceinline__ b16v frag2(const __bf16* p0, const __bf16* p1) {
    b8 x = *(const b8*)p0;
    b8 y = *(const b8*)p1;
    b16v r;
#pragma unroll
    for (int i = 0; i < 8; ++i) { r[i] = x[i]; r[i + 8] = y[i]; }
    return r;
}

// per-wave GEMM: 16 rows (in LDS, hi/lo bf16, stride KIN) x Wt[NT*16][KIN] -> C tiles
template<int KIN, int NT>
__device__ __forceinline__ void gemm_tile(const __bf16* AHi, const __bf16* ALo,
                                          const __bf16* __restrict__ WHi,
                                          const __bf16* __restrict__ WLo,
                                          int lane, v8f* C) {
    const int m    = lane & 15;
    const int half = lane >> 4;
    const int kb   = half * 8;
    v8f z = {0.f,0.f,0.f,0.f,0.f,0.f,0.f,0.f};
#pragma unroll
    for (int nt = 0; nt < NT; ++nt) C[nt] = z;
#pragma unroll
    for (int kt = 0; kt < KIN / 32; ++kt) {
        const __bf16* pa = AHi + m * KIN + kt * 32 + kb;
        const __bf16* pl = ALo + m * KIN + kt * 32 + kb;
        b16v aHi = frag2(pa, pa + 16);
        b16v aLo = frag2(pl, pl + 16);
#pragma unroll
        for (int nt = 0; nt < NT; ++nt) {
            const int n = nt * 16 + m;   // B column = lane&15
            const __bf16* pw = WHi + n * KIN + kt * 32 + half * 8;
            const __bf16* pv = WLo + n * KIN + kt * 32 + half * 8;
            b16v bHi = frag2(pw, pw + 16);
            b16v bLo = frag2(pv, pv + 16);
            C[nt] = __builtin_amdgcn_wmma_f32_16x16x32_bf16(false, aHi, false, bHi, (short)0, C[nt], false, false);
            C[nt] = __builtin_amdgcn_wmma_f32_16x16x32_bf16(false, aHi, false, bLo, (short)0, C[nt], false, false);
            C[nt] = __builtin_amdgcn_wmma_f32_16x16x32_bf16(false, aLo, false, bHi, (short)0, C[nt], false, false);
        }
    }
}

// bias + leaky-relu, re-split into next layer's LDS hi/lo activation buffers
template<int NOUT>
__device__ __forceinline__ void store_act(const v8f* C, const float* __restrict__ bias,
                                          __bf16* oHi, __bf16* oLo, int lane) {
    const int n0 = lane & 15, half = lane >> 4;
#pragma unroll
    for (int nt = 0; nt < NOUT / 16; ++nt) {
        float bv = bias[nt * 16 + n0];
#pragma unroll
        for (int r = 0; r < 8; ++r) {
            float v = C[nt][r] + bv;
            v = v > 0.f ? v : 0.01f * v;
            int idx = (r + 8 * half) * NOUT + nt * 16 + n0;
            unsigned short h, l; split2(v, h, l);
            oHi[idx] = us2bf(h);
            oLo[idx] = us2bf(l);
        }
    }
}

// final layer (64 wide): bias + leaky-relu, fp32 to LDS for LayerNorm
__device__ __forceinline__ void store_final(const v8f* C, const float* __restrict__ bias,
                                            float* o, int lane) {
    const int n0 = lane & 15, half = lane >> 4;
#pragma unroll
    for (int nt = 0; nt < 4; ++nt) {
        float bv = bias[nt * 16 + n0];
#pragma unroll
        for (int r = 0; r < 8; ++r) {
            float v = C[nt][r] + bv;
            v = v > 0.f ? v : 0.01f * v;
            o[(r + 8 * half) * 64 + nt * 16 + n0] = v;
        }
    }
}

__device__ __forceinline__ void layernorm16x64(const float* hf, float* mu, float* rstd, int lane) {
    wave_sync();
    if (lane < 16) {
        float s = 0.f;
        for (int c = 0; c < 64; ++c) s += hf[lane * 64 + c];
        float m = s * (1.0f / 64.0f);
        float s2 = 0.f;
        for (int c = 0; c < 64; ++c) { float d = hf[lane * 64 + c] - m; s2 += d * d; }
        mu[lane]   = m;
        rstd[lane] = rsqrtf(s2 * (1.0f / 64.0f) + 1e-5f);
    }
    wave_sync();
}

// ---------------- shared-memory slice (per wave) ----------------

struct alignas(64) WaveSmem {
    __bf16 aHi[2048];   // activations hi (16 rows x <=128), reused as float[1024] for LN
    __bf16 aLo[2048];   // activations lo
    int    rowIdx[16];
    int    colIdx[16];
    float  recip[16];
    float  mu[16];
    float  rstd[16];
};

// ---------------- kernels ----------------

__global__ void zero_f(float* __restrict__ p, int n) {
    int i = blockIdx.x * blockDim.x + threadIdx.x;
    if (i < n) p[i] = 0.f;
}

// split fp32 weight (K,N) row-major into transposed bf16 hi/lo [N][K]
__global__ void prep_w(const float* __restrict__ w, __bf16* __restrict__ hi,
                       __bf16* __restrict__ lo, int K, int Nout) {
    int i = blockIdx.x * blockDim.x + threadIdx.x;
    if (i >= K * Nout) return;
    int k = i / Nout, n = i - k * Nout;
    unsigned short h, l; split2(w[i], h, l);
    hi[n * K + k] = us2bf(h);
    lo[n * K + k] = us2bf(l);
}

// edge MLP: gather x[row]||edge_attr -> 96->128->128->64 -> LN -> atomic scatter to agg, count to cnt
__global__ void __launch_bounds__(128) edge_mlp(
    const float* __restrict__ x, const int* __restrict__ ei, const float* __restrict__ ea,
    const __bf16* __restrict__ w0h, const __bf16* __restrict__ w0l, const float* __restrict__ b0,
    const __bf16* __restrict__ w1h, const __bf16* __restrict__ w1l, const float* __restrict__ b1,
    const __bf16* __restrict__ w2h, const __bf16* __restrict__ w2l, const float* __restrict__ b2,
    const float* __restrict__ g, const float* __restrict__ bt,
    float* __restrict__ agg, float* __restrict__ cnt)
{
    __shared__ WaveSmem sm4[4];
    const int wv = threadIdx.x >> 5, lane = threadIdx.x & 31;
    WaveSmem& sm = sm4[wv];
    const int eBase = blockIdx.x * 64 + wv * 16;

    if (lane < 16) {
        int rw = ei[eBase + lane];
        int cl = ei[EE + eBase + lane];
        sm.rowIdx[lane] = rw;
        sm.colIdx[lane] = cl;
        unsafeAtomicAdd(&cnt[cl], 1.0f);
    }
    wave_sync();

    // stage concat(x[row], edge_attr) as bf16 hi/lo, 16 rows x 96
    for (int r = 0; r < 16; ++r) {
        int src = sm.rowIdx[r];
        int e = eBase + r;
        for (int c = lane; c < 96; c += 32) {
            float v = (c < 64) ? x[src * 64 + c] : ea[e * 32 + (c - 64)];
            unsigned short h, l; split2(v, h, l);
            sm.aHi[r * 96 + c] = us2bf(h);
            sm.aLo[r * 96 + c] = us2bf(l);
        }
    }

    v8f C[8];
    gemm_tile<96, 8>(sm.aHi, sm.aLo, w0h, w0l, lane, C);
    store_act<128>(C, b0, sm.aHi, sm.aLo, lane);
    gemm_tile<128, 8>(sm.aHi, sm.aLo, w1h, w1l, lane, C);
    store_act<128>(C, b1, sm.aHi, sm.aLo, lane);
    gemm_tile<128, 4>(sm.aHi, sm.aLo, w2h, w2l, lane, C);

    float* hf = (float*)sm.aHi;          // 16x64 fp32 reuses the hi buffer
    store_final(C, b2, hf, lane);
    layernorm16x64(hf, sm.mu, sm.rstd, lane);

    for (int i = lane; i < 1024; i += 32) {
        int r = i >> 6, c = i & 63;
        float v = (hf[i] - sm.mu[r]) * sm.rstd[r] * g[c] + bt[c];
        unsafeAtomicAdd(&agg[sm.colIdx[r] * 64 + c], v);
    }
}

// node MLP: concat(x, agg/cnt) -> 128->128->128->64 -> LN -> out
__global__ void __launch_bounds__(128) node_mlp(
    const float* __restrict__ x, const float* __restrict__ agg, const float* __restrict__ cnt,
    const __bf16* __restrict__ w0h, const __bf16* __restrict__ w0l, const float* __restrict__ b0,
    const __bf16* __restrict__ w1h, const __bf16* __restrict__ w1l, const float* __restrict__ b1,
    const __bf16* __restrict__ w2h, const __bf16* __restrict__ w2l, const float* __restrict__ b2,
    const float* __restrict__ g, const float* __restrict__ bt,
    float* __restrict__ out)
{
    __shared__ WaveSmem sm4[4];
    const int wv = threadIdx.x >> 5, lane = threadIdx.x & 31;
    WaveSmem& sm = sm4[wv];
    const int base = blockIdx.x * 64 + wv * 16;

    if (lane < 16) {
        int nr = base + lane;
        float c = (nr < NN) ? cnt[nr] : 1.0f;
        sm.recip[lane] = 1.0f / fmaxf(c, 1.0f);
    }
    wave_sync();

    for (int r = 0; r < 16; ++r) {
        int nr = base + r;
        bool ok = nr < NN;
        float rc = sm.recip[r];
        for (int c = lane; c < 128; c += 32) {
            float v = 0.f;
            if (ok) v = (c < 64) ? x[nr * 64 + c] : agg[nr * 64 + (c - 64)] * rc;
            unsigned short h, l; split2(v, h, l);
            sm.aHi[r * 128 + c] = us2bf(h);
            sm.aLo[r * 128 + c] = us2bf(l);
        }
    }

    v8f C[8];
    gemm_tile<128, 8>(sm.aHi, sm.aLo, w0h, w0l, lane, C);
    store_act<128>(C, b0, sm.aHi, sm.aLo, lane);
    gemm_tile<128, 8>(sm.aHi, sm.aLo, w1h, w1l, lane, C);
    store_act<128>(C, b1, sm.aHi, sm.aLo, lane);
    gemm_tile<128, 4>(sm.aHi, sm.aLo, w2h, w2l, lane, C);

    float* hf = (float*)sm.aHi;
    store_final(C, b2, hf, lane);
    layernorm16x64(hf, sm.mu, sm.rstd, lane);

    for (int i = lane; i < 1024; i += 32) {
        int r = i >> 6, c = i & 63;
        int nr = base + r;
        if (nr < NN) out[nr * 64 + c] = (hf[i] - sm.mu[r]) * sm.rstd[r] * g[c] + bt[c];
    }
}

// ---------------- launch ----------------

extern "C" void kernel_launch(void* const* d_in, const int* in_sizes, int n_in,
                              void* d_out, int out_size, void* d_ws, size_t ws_size,
                              hipStream_t stream) {
    const float* x    = (const float*)d_in[0];
    const int*   ei   = (const int*)d_in[1];
    const float* ea   = (const float*)d_in[2];
    // d_in[3]=u, d_in[4]=batch : unused
    const float* m1w0 = (const float*)d_in[5],  *m1b0 = (const float*)d_in[6];
    const float* m1w1 = (const float*)d_in[7],  *m1b1 = (const float*)d_in[8];
    const float* m1w2 = (const float*)d_in[9],  *m1b2 = (const float*)d_in[10];
    const float* g1   = (const float*)d_in[11], *bt1  = (const float*)d_in[12];
    const float* m2w0 = (const float*)d_in[13], *m2b0 = (const float*)d_in[14];
    const float* m2w1 = (const float*)d_in[15], *m2b1 = (const float*)d_in[16];
    const float* m2w2 = (const float*)d_in[17], *m2b2 = (const float*)d_in[18];
    const float* g2   = (const float*)d_in[19], *bt2  = (const float*)d_in[20];

    char* p = (char*)d_ws;
    float* agg = (float*)p; p += (size_t)NN * 64 * 4;
    float* cnt = (float*)p; p += (size_t)NN * 4;        // contiguous with agg
    auto alloc_bf = [&](size_t n) { __bf16* q = (__bf16*)p; p += n * 2; return q; };
    __bf16* w0h = alloc_bf(96 * 128),  *w0l = alloc_bf(96 * 128);
    __bf16* w1h = alloc_bf(128 * 128), *w1l = alloc_bf(128 * 128);
    __bf16* w2h = alloc_bf(128 * 64),  *w2l = alloc_bf(128 * 64);
    __bf16* v0h = alloc_bf(128 * 128), *v0l = alloc_bf(128 * 128);
    __bf16* v1h = alloc_bf(128 * 128), *v1l = alloc_bf(128 * 128);
    __bf16* v2h = alloc_bf(128 * 64),  *v2l = alloc_bf(128 * 64);

    zero_f<<<(NN * 65 + 255) / 256, 256, 0, stream>>>(agg, NN * 65);

    prep_w<<<(96 * 128 + 255) / 256, 256, 0, stream>>>(m1w0, w0h, w0l, 96, 128);
    prep_w<<<(128 * 128 + 255) / 256, 256, 0, stream>>>(m1w1, w1h, w1l, 128, 128);
    prep_w<<<(128 * 64 + 255) / 256, 256, 0, stream>>>(m1w2, w2h, w2l, 128, 64);
    prep_w<<<(128 * 128 + 255) / 256, 256, 0, stream>>>(m2w0, v0h, v0l, 128, 128);
    prep_w<<<(128 * 128 + 255) / 256, 256, 0, stream>>>(m2w1, v1h, v1l, 128, 128);
    prep_w<<<(128 * 64 + 255) / 256, 256, 0, stream>>>(m2w2, v2h, v2l, 128, 64);

    edge_mlp<<<EE / 64, 128, 0, stream>>>(x, ei, ea,
        w0h, w0l, m1b0, w1h, w1l, m1b1, w2h, w2l, m1b2, g1, bt1, agg, cnt);

    node_mlp<<<(NN + 63) / 64, 128, 0, stream>>>(x, agg, cnt,
        v0h, v0l, m2b0, v1h, v1l, m2b1, v2h, v2l, m2b2, g2, bt2, (float*)d_out);
}
